// DeepFusedGPTTransformerLayer_77850577207483
// MI455X (gfx1250) — matrix operand order repaired and checked by reading, not compiled, not run on hardware
//
#include <hip/hip_runtime.h>
#include <hip/hip_bf16.h>

typedef __bf16 bf16_t;
typedef __attribute__((ext_vector_type(16))) __bf16 v16bf;
typedef __attribute__((ext_vector_type(8)))  float  v8f;

#define B_  2
#define S_  2048
#define H_  2048
#define NH_ 16
#define HD_ 128
#define FF_ 8192
#define M_  (B_ * S_)
#define SCALE_ 0.08838834764831845f   // 1/sqrt(128)

// ---------------- CDNA5 async global->LDS DMA (ASYNCcnt-tracked) ----------------
static __device__ __forceinline__ void async_copy_b128(void* lds_ptr, const void* gptr) {
  const unsigned lds_off = (unsigned)(size_t)lds_ptr;   // low 32b of generic LDS addr = LDS byte offset
  asm volatile("global_load_async_to_lds_b128 %0, %1, off"
               :: "v"(lds_off), "v"((unsigned long long)(size_t)gptr)
               : "memory");
}
static __device__ __forceinline__ void wait_async0() {
  asm volatile("s_wait_asynccnt 0" ::: "memory");
}

// ---------------- WMMA fragment helpers (CDNA5 16x16x32 bf16 layouts) ----------------
// A-matrix 16x32: lane<16 holds row (lane), K = {0..7, 16..23}; lane>=16 holds K = {8..15, 24..31}
static __device__ __forceinline__ v16bf ld_frag_a(const bf16_t* base, int stride, int lane) {
  const bf16_t* p = base + (lane & 15) * stride + ((lane < 16) ? 0 : 8);
  union { v16bf v; uint4 u[2]; } t;
  t.u[0] = *(const uint4*)(p);
  t.u[1] = *(const uint4*)(p + 16);
  return t.v;
}
// B-matrix 32x16 stored column-major-in-LDS ([n][k], row stride = `stride`):
// lane n<16 holds col n, K = {0..7, 16..23}; lane>=16 holds col n-16, K = {8..15, 24..31}
static __device__ __forceinline__ v16bf ld_frag_b(const bf16_t* base, int stride, int lane) {
  const bf16_t* p = base + (lane & 15) * stride + ((lane < 16) ? 0 : 8);
  union { v16bf v; uint4 u[2]; } t;
  t.u[0] = *(const uint4*)(p);
  t.u[1] = *(const uint4*)(p + 16);
  return t.v;
}
static __device__ __forceinline__ v8f wmma_bf16(v16bf a, v16bf b, v8f c) {
  return __builtin_amdgcn_wmma_f32_16x16x32_bf16(false, a, false, b, (short)0, c, false, false);
}

// ---------------- LayerNorm: fp32 [rows, H_] -> bf16 ----------------
__global__ __launch_bounds__(256) void ln_bf16_kernel(
    const float* __restrict__ x, const float* __restrict__ g,
    const float* __restrict__ bta, bf16_t* __restrict__ y) {
  const int row = blockIdx.x;
  const float* xr = x + (size_t)row * H_;
  const int tid = threadIdx.x, lane = tid & 31, wave = tid >> 5;
  float s = 0.f, s2 = 0.f;
  #pragma unroll
  for (int i = tid; i < H_; i += 256) { float v = xr[i]; s += v; s2 += v * v; }
  #pragma unroll
  for (int off = 16; off >= 1; off >>= 1) {
    s  += __shfl_xor(s,  off, 32);
    s2 += __shfl_xor(s2, off, 32);
  }
  __shared__ float sm[16];
  __shared__ float stats[2];
  if (lane == 0) { sm[wave] = s; sm[8 + wave] = s2; }
  __syncthreads();
  if (tid == 0) {
    float a = 0.f, b2 = 0.f;
    for (int w = 0; w < 8; ++w) { a += sm[w]; b2 += sm[8 + w]; }
    const float mean = a / (float)H_;
    stats[0] = mean;
    stats[1] = b2 / (float)H_ - mean * mean;
  }
  __syncthreads();
  const float mean = stats[0];
  const float inv  = rsqrtf(stats[1] + 1e-5f);
  bf16_t* yr = y + (size_t)row * H_;
  #pragma unroll
  for (int i = tid; i < H_; i += 256)
    yr[i] = (bf16_t)((xr[i] - mean) * inv * g[i] + bta[i]);
}

// ---------------- GEMM: C[M,N] = A_bf16[M,K] @ W_f32[K,N] (+bias, epilogue) ----------------
// EPI 0: bias -> bf16 out;  EPI 1: bias + exact GELU -> bf16 out;  EPI 2: bias + residual -> f32 out
template <int EPI>
__global__ __launch_bounds__(256) void gemm_bf16_kernel(
    const bf16_t* __restrict__ A, const float* __restrict__ W,
    const float* __restrict__ bias, const float* __restrict__ res,
    void* __restrict__ outp, int M, int N, int K) {
  __shared__ alignas(16) bf16_t lA[128 * 32];   // [m][k] row-major
  __shared__ alignas(16) bf16_t lB[128 * 32];   // [n][k] (transposed during store)
  const int tid = threadIdx.x, lane = tid & 31, wave = tid >> 5;
  const int wm = wave >> 1, wn = wave & 1;                  // 4x2 wave grid
  const int bm = blockIdx.y * 128, bn = blockIdx.x * 128;

  v8f acc[2][4];
  #pragma unroll
  for (int mi = 0; mi < 2; ++mi)
    #pragma unroll
    for (int ni = 0; ni < 4; ++ni)
      #pragma unroll
      for (int e = 0; e < 8; ++e) acc[mi][ni][e] = 0.f;

  const int ra = tid >> 1, ca = (tid & 1) * 16;             // A tile: 128 rows x 32
  const int rb = tid >> 3, cb = (tid & 7) * 16;             // W tile: 32 rows x 128

  for (int k0 = 0; k0 < K; k0 += 32) {
    __syncthreads();
    {   // stage A (bf16, row-major): async DMA straight into LDS, 32 B per thread
      const bf16_t* src = A + (size_t)(bm + ra) * K + k0 + ca;
      bf16_t* dst = lA + ra * 32 + ca;
      async_copy_b128(dst, src);
      async_copy_b128(dst + 8, src + 8);
    }
    {   // stage W (fp32 -> bf16), transpose into [n][k] (needs VALU conversion)
      const float4* s4 = (const float4*)(W + (size_t)(k0 + rb) * N + bn + cb);
      #pragma unroll
      for (int v = 0; v < 4; ++v) {
        float4 f = s4[v];
        const int c = cb + v * 4;
        lB[(c + 0) * 32 + rb] = (bf16_t)f.x;
        lB[(c + 1) * 32 + rb] = (bf16_t)f.y;
        lB[(c + 2) * 32 + rb] = (bf16_t)f.z;
        lB[(c + 3) * 32 + rb] = (bf16_t)f.w;
      }
    }
    wait_async0();
    __syncthreads();
    v16bf af[2];
    #pragma unroll
    for (int mi = 0; mi < 2; ++mi)
      af[mi] = ld_frag_a(lA + (wm * 32 + mi * 16) * 32, 32, lane);
    #pragma unroll
    for (int ni = 0; ni < 4; ++ni) {
      v16bf bfr = ld_frag_b(lB + (wn * 64 + ni * 16) * 32, 32, lane);
      #pragma unroll
      for (int mi = 0; mi < 2; ++mi)
        acc[mi][ni] = wmma_bf16(af[mi], bfr, acc[mi][ni]);
    }
  }

  const int rbase = (lane < 16) ? 0 : 8;
  #pragma unroll
  for (int mi = 0; mi < 2; ++mi) {
    #pragma unroll
    for (int ni = 0; ni < 4; ++ni) {
      const int n  = bn + wn * 64 + ni * 16 + (lane & 15);
      const int m0 = bm + wm * 32 + mi * 16 + rbase;
      const float bs = bias[n];
      #pragma unroll
      for (int r = 0; r < 8; ++r) {
        const size_t idx = (size_t)(m0 + r) * N + n;
        float v = acc[mi][ni][r] + bs;
        if (EPI == 1) v = 0.5f * v * (1.f + erff(v * 0.70710678118654752f));
        if (EPI == 2) {
          ((float*)outp)[idx] = v + res[idx];
        } else {
          ((bf16_t*)outp)[idx] = (bf16_t)v;
        }
      }
    }
  }
}

// ---------------- RoPE + head split: qkv bf16 [B,S,NH,3,HD] -> q,k [B,NH,S,HD], v^T [B,NH,HD,S] ----------------
__global__ __launch_bounds__(128) void rope_split_kernel(
    const bf16_t* __restrict__ qkv, bf16_t* __restrict__ q,
    bf16_t* __restrict__ k, bf16_t* __restrict__ vt) {
  const int s = blockIdx.x, h = blockIdx.y, b = blockIdx.z, d = threadIdx.x;
  const bf16_t* base = qkv + (size_t)(b * S_ + s) * (NH_ * 3 * HD_) + h * (3 * HD_);
  const float qv = (float)base[d];
  const float kv = (float)base[HD_ + d];
  const float vv = (float)base[2 * HD_ + d];
  const int   dp = (d < 64) ? d + 64 : d - 64;
  const float sg = (d < 64) ? -1.f : 1.f;
  const float qp = (float)base[dp];
  const float kp = (float)base[HD_ + dp];
  // inv_freq = 10000^(-(d%64)/64)
  const float invf = __expf(-(float)(d & 63) * (9.210340371976184f / 64.f));
  float c, sn;
  __sincosf((float)s * invf, &c, &sn);
  const size_t off = ((size_t)(b * NH_ + h) * S_ + s) * HD_ + d;
  q[off] = (bf16_t)(qv * c + sg * qp * sn);
  k[off] = (bf16_t)(kv * c + sg * kp * sn);
  vt[((size_t)(b * NH_ + h) * HD_ + d) * S_ + s] = (bf16_t)vv;
}

// ---------------- Flash attention (causal), WMMA QK^T and P.V, online softmax ----------------
__global__ __launch_bounds__(256) void flash_attn_kernel(
    const bf16_t* __restrict__ Q, const bf16_t* __restrict__ Kk,
    const bf16_t* __restrict__ Vt, bf16_t* __restrict__ ctx) {
  __shared__ alignas(16) bf16_t lK[32 * HD_];      // [key][d]
  __shared__ alignas(16) bf16_t lV[HD_ * 32];      // [d][key]  (V already transposed in global)
  __shared__ alignas(16) bf16_t lP[8 * 16 * 32];   // per-wave P tile [16 q][32 key]
  const int qblk = blockIdx.x, h = blockIdx.y, b = blockIdx.z;
  const int tid = threadIdx.x, lane = tid & 31, wave = tid >> 5;
  const int q0 = qblk * 128 + wave * 16;
  const bf16_t* Qb = Q  + (size_t)(b * NH_ + h) * S_ * HD_;
  const bf16_t* Kb = Kk + (size_t)(b * NH_ + h) * S_ * HD_;
  const bf16_t* Vb = Vt + (size_t)(b * NH_ + h) * HD_ * S_;

  v16bf qf[4];
  #pragma unroll
  for (int j = 0; j < 4; ++j)
    qf[j] = ld_frag_a(Qb + (size_t)q0 * HD_ + j * 32, HD_, lane);

  float m_i[8], l_i[8];
  v8f cacc[8];
  #pragma unroll
  for (int r = 0; r < 8; ++r) { m_i[r] = -3.0e38f; l_i[r] = 0.f; }
  #pragma unroll
  for (int dt = 0; dt < 8; ++dt)
    #pragma unroll
    for (int e = 0; e < 8; ++e) cacc[dt][e] = 0.f;

  const int kend = (qblk + 1) * 128;
  const int rK = tid >> 3, cK = (tid & 7) * 16;   // K tile loader: [32][128]
  const int rV = tid >> 1, cV = (tid & 1) * 16;   // V tile loader: [128][32]
  bf16_t* Pw = lP + wave * (16 * 32);

  for (int kb = 0; kb < kend; kb += 32) {
    __syncthreads();
    {   // stage K block [32][128] via async DMA
      const bf16_t* src = Kb + (size_t)(kb + rK) * HD_ + cK;
      bf16_t* dst = lK + rK * HD_ + cK;
      async_copy_b128(dst, src);
      async_copy_b128(dst + 8, src + 8);
    }
    {   // stage V^T block [128][32] via async DMA
      const bf16_t* src = Vb + (size_t)rV * S_ + kb + cV;
      bf16_t* dst = lV + rV * 32 + cV;
      async_copy_b128(dst, src);
      async_copy_b128(dst + 8, src + 8);
    }
    wait_async0();
    __syncthreads();

    // scores: two 16x16 tiles (keys kb..+15, kb+16..+31), K-dim = d in 4 chunks of 32
    v8f s0, s1;
    #pragma unroll
    for (int e = 0; e < 8; ++e) { s0[e] = 0.f; s1[e] = 0.f; }
    #pragma unroll
    for (int j = 0; j < 4; ++j) {
      v16bf b0 = ld_frag_b(lK + j * 32,            HD_, lane);
      v16bf b1 = ld_frag_b(lK + 16 * HD_ + j * 32, HD_, lane);
      s0 = wmma_bf16(qf[j], b0, s0);
      s1 = wmma_bf16(qf[j], b1, s1);
    }

    // online softmax per q-row (C layout: VGPR r -> row r + 8*(lane>=16); lane%16 -> column)
    float alpha[8];
    #pragma unroll
    for (int r = 0; r < 8; ++r) {
      const int qg  = q0 + r + ((lane < 16) ? 0 : 8);
      const int k0i = kb + (lane & 15);
      float a0 = s0[r] * SCALE_; if (k0i      > qg) a0 = -1.0e30f;
      float a1 = s1[r] * SCALE_; if (k0i + 16 > qg) a1 = -1.0e30f;
      float mx = fmaxf(a0, a1);
      #pragma unroll
      for (int off = 8; off >= 1; off >>= 1) mx = fmaxf(mx, __shfl_xor(mx, off, 32));
      const float mn = fmaxf(m_i[r], mx);
      const float al = __expf(m_i[r] - mn);
      const float p0 = __expf(a0 - mn);
      const float p1 = __expf(a1 - mn);
      float sum = p0 + p1;
      #pragma unroll
      for (int off = 8; off >= 1; off >>= 1) sum += __shfl_xor(sum, off, 32);
      l_i[r] = l_i[r] * al + sum;
      m_i[r] = mn;
      alpha[r] = al;
      const int row = r + ((lane < 16) ? 0 : 8);
      Pw[row * 32 +      (lane & 15)] = (bf16_t)p0;
      Pw[row * 32 + 16 + (lane & 15)] = (bf16_t)p1;
    }
    #pragma unroll
    for (int dt = 0; dt < 8; ++dt)
      #pragma unroll
      for (int r = 0; r < 8; ++r) cacc[dt][r] *= alpha[r];

    asm volatile("s_wait_dscnt 0" ::: "memory");   // P store -> P re-load (same wave, cross-lane)
    __builtin_amdgcn_wave_barrier();

    v16bf pf = ld_frag_a(Pw, 32, lane);            // C-layout -> A-layout via LDS
    #pragma unroll
    for (int dt = 0; dt < 8; ++dt) {
      v16bf vfr = ld_frag_b(lV + dt * 16 * 32, 32, lane);
      cacc[dt] = wmma_bf16(pf, vfr, cacc[dt]);
    }
  }

  // normalize and scatter ctx into [B,S,H] (bf16) for the dense GEMM
  bf16_t* ob = ctx + (size_t)b * S_ * H_ + h * HD_;
  #pragma unroll
  for (int dt = 0; dt < 8; ++dt) {
    const int d = dt * 16 + (lane & 15);
    #pragma unroll
    for (int r = 0; r < 8; ++r) {
      const int srow = q0 + r + ((lane < 16) ? 0 : 8);
      ob[(size_t)srow * H_ + d] = (bf16_t)(cacc[dt][r] / l_i[r]);
    }
  }
}

// ---------------- launch ----------------
extern "C" void kernel_launch(void* const* d_in, const int* in_sizes, int n_in,
                              void* d_out, int out_size, void* d_ws, size_t ws_size,
                              hipStream_t stream) {
  (void)in_sizes; (void)n_in; (void)out_size; (void)ws_size;
  const float* hs      = (const float*)d_in[0];
  // d_in[1] attention_mask: all-false, unused (reference applies causal mask internally)
  const float* ln1_g   = (const float*)d_in[2];
  const float* ln1_b   = (const float*)d_in[3];
  const float* w_qkv   = (const float*)d_in[4];
  const float* b_qkv   = (const float*)d_in[5];
  const float* w_dense = (const float*)d_in[6];
  const float* b_dense = (const float*)d_in[7];
  const float* ln2_g   = (const float*)d_in[8];
  const float* ln2_b   = (const float*)d_in[9];
  const float* w1      = (const float*)d_in[10];
  const float* b1      = (const float*)d_in[11];
  const float* w2      = (const float*)d_in[12];
  const float* b2      = (const float*)d_in[13];

  // workspace layout (peak 128 MB, regions reused once dead):
  char* ws = (char*)d_ws;
  bf16_t* xln    = (bf16_t*)(ws);                    // 16 MB  [0,16)
  bf16_t* qkv    = (bf16_t*)(ws + (16ull << 20));    // 48 MB  [16,64)
  bf16_t* qm     = (bf16_t*)(ws + (64ull << 20));    // 16 MB  [64,80)
  bf16_t* km     = (bf16_t*)(ws + (80ull << 20));    // 16 MB  [80,96)
  bf16_t* vtm    = (bf16_t*)(ws + (96ull << 20));    // 16 MB  [96,112)
  bf16_t* ctx    = (bf16_t*)(ws + (112ull << 20));   // 16 MB  [112,128)
  float*  hidden = (float*)(ws + (64ull << 20));     // 32 MB  [64,96)  (q/k dead after flash)
  bf16_t* yln    = (bf16_t*)(ws + (96ull << 20));    // 16 MB  [96,112) (v^T dead after flash)
  bf16_t* h1     = (bf16_t*)(ws);                    // 64 MB  [0,64)   (xln/qkv dead)

  ln_bf16_kernel<<<M_, 256, 0, stream>>>(hs, ln1_g, ln1_b, xln);
  gemm_bf16_kernel<0><<<dim3(3 * H_ / 128, M_ / 128), 256, 0, stream>>>(
      xln, w_qkv, b_qkv, nullptr, qkv, M_, 3 * H_, H_);
  rope_split_kernel<<<dim3(S_, NH_, B_), 128, 0, stream>>>(qkv, qm, km, vtm);
  flash_attn_kernel<<<dim3(S_ / 128, NH_, B_), 256, 0, stream>>>(qm, km, vtm, ctx);
  gemm_bf16_kernel<2><<<dim3(H_ / 128, M_ / 128), 256, 0, stream>>>(
      ctx, w_dense, b_dense, hs, hidden, M_, H_, H_);
  ln_bf16_kernel<<<M_, 256, 0, stream>>>(hidden, ln2_g, ln2_b, yln);
  gemm_bf16_kernel<1><<<dim3(FF_ / 128, M_ / 128), 256, 0, stream>>>(
      yln, w1, b1, nullptr, h1, M_, FF_, H_);
  gemm_bf16_kernel<2><<<dim3(H_ / 128, M_ / 128), 256, 0, stream>>>(
      h1, w2, b2, hidden, (float*)d_out, M_, H_, FF_);
}
